// BinaryLinear_31353261261103
// MI455X (gfx1250) — compile-verified, matrix operand order checked by reading, not run on hardware
//
#include <hip/hip_runtime.h>
#include <hip/hip_bf16.h>

// ---------------------------------------------------------------------------
// BinaryLinear: y = x @ sign(W)^T + sign(bias)
//   x[8192,4096] f32, W[4096,4096] f32, bias[4096] f32 -> y[8192,4096] f32
// Compute-bound GEMM (274.9 GFLOP vs ~14us HBM floor).
// Phase 1: one-shot fp32->bf16 (x) and fp32->sign-bf16 (W) into d_ws —
//   removes the 32-64x redundant per-tile conversion VALU from the GEMM loop
//   and halves GEMM global read traffic.
// Phase 2: bf16 WMMA GEMM, BK=64, double-buffered LDS fed by CDNA5 async
//   global->LDS copies (ASYNCcnt), 16x v_wmma_f32_16x16x32_bf16 per wave/step.
// Fallback (ws too small): inline-conversion variants.
// ---------------------------------------------------------------------------

typedef __attribute__((ext_vector_type(16))) __bf16 v16bf;
typedef __attribute__((ext_vector_type(8)))  float  v8f;
typedef __attribute__((ext_vector_type(4)))  int    v4i;

#define TOKENS 8192
#define INDIM  4096
#define OUTDIM 4096

union FragU {
    v16bf v;
    uint4 q[2];
};

__device__ __forceinline__ uint32_t bf16_rne(float f) {
    uint32_t u = __float_as_uint(f);
    return (u + 0x7FFFu + ((u >> 16) & 1u)) >> 16;
}
__device__ __forceinline__ uint32_t pack_bf16(float lo, float hi) {
    return bf16_rne(lo) | (bf16_rne(hi) << 16);
}
// sign(w) as packed bf16: +1.0 (0x3F80) or -1.0 (0xBF80) from fp32 sign bit
__device__ __forceinline__ uint32_t pack_sgn(uint32_t a, uint32_t b) {
    return (0x3F80u | ((a >> 16) & 0x8000u)) |
           ((0x3F80u | ((b >> 16) & 0x8000u)) << 16);
}

// ---------------------------------------------------------------------------
// CDNA5 async global->LDS copy (tracked by ASYNCcnt)
// Builtin signature (per hipcc diagnostic): (v4i AS1*, v4i AS3*, Ii, Ii)
// ---------------------------------------------------------------------------
#if __has_builtin(__builtin_amdgcn_global_load_async_to_lds_b128)
#define HAVE_ASYNC_LDS 1
typedef __attribute__((address_space(1))) v4i as1_v4i;
typedef __attribute__((address_space(3))) v4i as3_v4i;
#endif

__device__ __forceinline__ void async_copy16(const void* g, void* l) {
#ifdef HAVE_ASYNC_LDS
    __builtin_amdgcn_global_load_async_to_lds_b128(
        (as1_v4i*)g, (as3_v4i*)l, 0, 0);
#else
    *(uint4*)l = *(const uint4*)g;
#endif
}
__device__ __forceinline__ void stage_fence() {
#ifdef HAVE_ASYNC_LDS
#if __has_builtin(__builtin_amdgcn_s_wait_asynccnt)
    __builtin_amdgcn_s_wait_asynccnt(0);
#else
    asm volatile("s_wait_asynccnt 0x0" ::: "memory");
#endif
#endif
}

// ---------------------------------------------------------------------------
// Phase 1: conversion kernels (flat, no loop, 8 f32 -> 16B bf16 / thread)
// ---------------------------------------------------------------------------
__global__ __launch_bounds__(256)
void prep_x_kernel(const float* __restrict__ x, uint32_t* __restrict__ dst) {
    size_t i = ((size_t)blockIdx.x * 256 + threadIdx.x) * 8;
    float4 a = *(const float4*)(x + i);
    float4 b = *(const float4*)(x + i + 4);
    uint4 o;
    o.x = pack_bf16(a.x, a.y);  o.y = pack_bf16(a.z, a.w);
    o.z = pack_bf16(b.x, b.y);  o.w = pack_bf16(b.z, b.w);
    *(uint4*)(dst + i / 2) = o;
}

__global__ __launch_bounds__(256)
void prep_w_kernel(const uint32_t* __restrict__ w, uint32_t* __restrict__ dst) {
    size_t i = ((size_t)blockIdx.x * 256 + threadIdx.x) * 8;
    uint4 a = *(const uint4*)(w + i);
    uint4 b = *(const uint4*)(w + i + 4);
    uint4 o;
    o.x = pack_sgn(a.x, a.y);  o.y = pack_sgn(a.z, a.w);
    o.z = pack_sgn(b.x, b.y);  o.w = pack_sgn(b.z, b.w);
    *(uint4*)(dst + i / 2) = o;
}

// ---------------------------------------------------------------------------
// Phase 2: bf16 WMMA GEMM from pre-converted operands
//   128x128 block tile, BK=64, 8 waves (4M x 2N), 32x64 per wave.
// ---------------------------------------------------------------------------
#define BM 128
#define BN 128
#define BK2 64
#define NK2 (INDIM / BK2)   // 64 K-steps
#define LDSP2 72            // pitch: 64 bf16 + 8 pad -> 144B = 36 banks

__global__ __launch_bounds__(256, 2)
void binlin_gemm_bf16(const __bf16* __restrict__ A,   // x      [M][K] bf16
                      const __bf16* __restrict__ B,   // sgn(W) [N][K] bf16
                      const float* __restrict__ bias,
                      float* __restrict__ out) {
    __shared__ __align__(16) __bf16 Asb[2][BM * LDSP2];
    __shared__ __align__(16) __bf16 Bsb[2][BN * LDSP2];

    const int tid  = threadIdx.x;
    const int lane = tid & 31;
    const int wave = tid >> 5;

    // block swizzle: supertiles of 8 M-tiles for L2 residency
    const int num_pid_m = TOKENS / BM;   // 64
    const int num_pid_n = OUTDIM / BN;   // 32
    const int GROUP = 8;
    int pid = blockIdx.x;
    int npg = GROUP * num_pid_n;
    int gid = pid / npg;
    int first_m = gid * GROUP;
    int gsz = num_pid_m - first_m; if (gsz > GROUP) gsz = GROUP;
    const int row0 = (first_m + (pid % npg) % gsz) * BM;
    const int col0 = ((pid % npg) / gsz) * BN;

    // staging: 8 lanes x 16B cover one 128B row; 32 rows per pass, 4 passes
    const int lr  = tid >> 3;          // 0..31
    const int lch = (tid & 7) * 8;     // bf16 col 0,8,..,56

    // wave tile mapping
    const int waveM = wave >> 1;       // rows waveM*32
    const int waveN = wave & 1;        // cols waveN*64
    const int fr    = lane & 15;       // fragment row (M for A, N for B)
    const int kh    = (lane >> 4) << 3;    // K half-select 0/8 (ISA layout)

    v8f acc[2][4] = {};

    // async-stage one K-step tile pair into LDS buffer `buf`
    auto stage = [&](int buf, int k0) {
        #pragma unroll
        for (int p = 0; p < 4; ++p) {
            const int r = lr + p * 32;
            async_copy16(A + (size_t)(row0 + r) * INDIM + k0 + lch,
                         &Asb[buf][r * LDSP2 + lch]);
            async_copy16(B + (size_t)(col0 + r) * INDIM + k0 + lch,
                         &Bsb[buf][r * LDSP2 + lch]);
        }
    };

    stage(0, 0);
    stage_fence();
    __syncthreads();

    for (int k = 0; k < NK2; ++k) {
        const int cur = k & 1;

        // prefetch next K-step into the other buffer (hidden by 16 WMMAs)
        if (k + 1 < NK2) stage(cur ^ 1, (k + 1) * BK2);

        // two k-subtiles of 32 within BK=64
        #pragma unroll
        for (int ks = 0; ks < 2; ++ks) {
            const int kb = ks * 32 + kh;   // bf16 elem offset within LDS row

            // fragments per ISA 16-bit A 16x32 layout:
            // lanes 0-15: K 0-7 & 16-23 ; lanes 16-31: K 8-15 & 24-31
            FragU a[2], b[4];
            #pragma unroll
            for (int mi = 0; mi < 2; ++mi) {
                const __bf16* p = &Asb[cur][(waveM * 32 + mi * 16 + fr) * LDSP2 + kb];
                a[mi].q[0] = *(const uint4*)(p);
                a[mi].q[1] = *(const uint4*)(p + 16);
            }
            #pragma unroll
            for (int ni = 0; ni < 4; ++ni) {
                const __bf16* p = &Bsb[cur][(waveN * 64 + ni * 16 + fr) * LDSP2 + kb];
                b[ni].q[0] = *(const uint4*)(p);
                b[ni].q[1] = *(const uint4*)(p + 16);
            }

            #pragma unroll
            for (int mi = 0; mi < 2; ++mi)
                #pragma unroll
                for (int ni = 0; ni < 4; ++ni)
                    acc[mi][ni] = __builtin_amdgcn_wmma_f32_16x16x32_bf16(
                        false, a[mi].v, false, b[ni].v,
                        (short)0, acc[mi][ni], false, false);
        }

        stage_fence();     // this wave's async copies for buf^1 complete
        __syncthreads();   // all waves' copies visible
    }

    // epilogue: D layout (VGPR r -> row r / r+8; lane -> col), add sign(bias)
    const int rsel = (lane >> 4) << 3;
    #pragma unroll
    for (int ni = 0; ni < 4; ++ni) {
        const int col = col0 + waveN * 64 + ni * 16 + fr;
        const float bs = (bias[col] >= 0.0f) ? 1.0f : -1.0f;
        #pragma unroll
        for (int mi = 0; mi < 2; ++mi) {
            const int rbase = row0 + waveM * 32 + mi * 16 + rsel;
            #pragma unroll
            for (int r = 0; r < 8; ++r)
                out[(size_t)(rbase + r) * OUTDIM + col] = acc[mi][ni][r] + bs;
        }
    }
}

// ---------------------------------------------------------------------------
// Fallback: inline-conversion GEMM (used if d_ws cannot hold bf16 operands)
// ---------------------------------------------------------------------------
#define BK 32
#define NK (INDIM / BK)
#define LDSP 40

__global__ __launch_bounds__(256, 2)
void binlin_wmma_fallback(const float* __restrict__ x,
                          const float* __restrict__ w,
                          const float* __restrict__ bias,
                          float* __restrict__ out) {
    __shared__ __align__(16) __bf16 Asb[2][BM * LDSP];
    __shared__ __align__(16) __bf16 Bsb[2][BN * LDSP];

    const int tid  = threadIdx.x;
    const int lane = tid & 31;
    const int wave = tid >> 5;

    const int num_pid_m = TOKENS / BM;
    const int num_pid_n = OUTDIM / BN;
    const int GROUP = 8;
    int pid = blockIdx.x;
    int npg = GROUP * num_pid_n;
    int gid = pid / npg;
    int first_m = gid * GROUP;
    int gsz = num_pid_m - first_m; if (gsz > GROUP) gsz = GROUP;
    const int row0 = (first_m + (pid % npg) % gsz) * BM;
    const int col0 = ((pid % npg) / gsz) * BN;

    const int lr = tid >> 3;
    const int lc = (tid & 7) << 2;

    const int waveM = wave >> 1;
    const int waveN = wave & 1;
    const int fr    = lane & 15;
    const int kh    = (lane >> 4) << 3;

    v8f acc[2][4] = {};
    float4 xr[4];
    uint4  wr[4];

    #pragma unroll
    for (int p = 0; p < 4; ++p) {
        xr[p] = *(const float4*)(x + (size_t)(row0 + lr + p * 32) * INDIM + lc);
        wr[p] = *(const uint4*)(w + (size_t)(col0 + lr + p * 32) * INDIM + lc);
    }
    #pragma unroll
    for (int p = 0; p < 4; ++p) {
        uint2 pa, pb;
        pa.x = pack_bf16(xr[p].x, xr[p].y);
        pa.y = pack_bf16(xr[p].z, xr[p].w);
        pb.x = pack_sgn(wr[p].x, wr[p].y);
        pb.y = pack_sgn(wr[p].z, wr[p].w);
        *(uint2*)(&Asb[0][(lr + p * 32) * LDSP + lc]) = pa;
        *(uint2*)(&Bsb[0][(lr + p * 32) * LDSP + lc]) = pb;
    }
    __syncthreads();

    for (int k = 0; k < NK; ++k) {
        const int cur = k & 1;
        if (k + 1 < NK) {
            const int kn = (k + 1) * BK;
            #pragma unroll
            for (int p = 0; p < 4; ++p) {
                xr[p] = *(const float4*)(x + (size_t)(row0 + lr + p * 32) * INDIM + kn + lc);
                wr[p] = *(const uint4*)(w + (size_t)(col0 + lr + p * 32) * INDIM + kn + lc);
            }
        }

        FragU a[2], b[4];
        #pragma unroll
        for (int mi = 0; mi < 2; ++mi) {
            const __bf16* p = &Asb[cur][(waveM * 32 + mi * 16 + fr) * LDSP + kh];
            a[mi].q[0] = *(const uint4*)(p);
            a[mi].q[1] = *(const uint4*)(p + 16);
        }
        #pragma unroll
        for (int ni = 0; ni < 4; ++ni) {
            const __bf16* p = &Bsb[cur][(waveN * 64 + ni * 16 + fr) * LDSP + kh];
            b[ni].q[0] = *(const uint4*)(p);
            b[ni].q[1] = *(const uint4*)(p + 16);
        }
        #pragma unroll
        for (int mi = 0; mi < 2; ++mi)
            #pragma unroll
            for (int ni = 0; ni < 4; ++ni)
                acc[mi][ni] = __builtin_amdgcn_wmma_f32_16x16x32_bf16(
                    false, a[mi].v, false, b[ni].v,
                    (short)0, acc[mi][ni], false, false);

        if (k + 1 < NK) {
            const int nxt = cur ^ 1;
            #pragma unroll
            for (int p = 0; p < 4; ++p) {
                uint2 pa, pb;
                pa.x = pack_bf16(xr[p].x, xr[p].y);
                pa.y = pack_bf16(xr[p].z, xr[p].w);
                pb.x = pack_sgn(wr[p].x, wr[p].y);
                pb.y = pack_sgn(wr[p].z, wr[p].w);
                *(uint2*)(&Asb[nxt][(lr + p * 32) * LDSP + lc]) = pa;
                *(uint2*)(&Bsb[nxt][(lr + p * 32) * LDSP + lc]) = pb;
            }
        }
        __syncthreads();
    }

    const int rsel = (lane >> 4) << 3;
    #pragma unroll
    for (int ni = 0; ni < 4; ++ni) {
        const int col = col0 + waveN * 64 + ni * 16 + fr;
        const float bs = (bias[col] >= 0.0f) ? 1.0f : -1.0f;
        #pragma unroll
        for (int mi = 0; mi < 2; ++mi) {
            const int rbase = row0 + waveM * 32 + mi * 16 + rsel;
            #pragma unroll
            for (int r = 0; r < 8; ++r)
                out[(size_t)(rbase + r) * OUTDIM + col] = acc[mi][ni][r] + bs;
        }
    }
}

// ---------------------------------------------------------------------------
extern "C" void kernel_launch(void* const* d_in, const int* in_sizes, int n_in,
                              void* d_out, int out_size, void* d_ws, size_t ws_size,
                              hipStream_t stream) {
    const float* x    = (const float*)d_in[0];
    const float* w    = (const float*)d_in[1];
    const float* bias = (const float*)d_in[2];
    float* out        = (float*)d_out;

    const size_t xbytes = (size_t)TOKENS * INDIM * sizeof(__bf16);   // 64 MiB
    const size_t wbytes = (size_t)OUTDIM * INDIM * sizeof(__bf16);   // 32 MiB
    const int grid = (TOKENS / BM) * (OUTDIM / BN);                  // 2048

    if (ws_size >= xbytes + wbytes) {
        uint32_t* xa = (uint32_t*)d_ws;
        uint32_t* wa = (uint32_t*)((char*)d_ws + xbytes);
        prep_x_kernel<<<(TOKENS * INDIM) / (8 * 256), 256, 0, stream>>>(x, xa);
        prep_w_kernel<<<(OUTDIM * INDIM) / (8 * 256), 256, 0, stream>>>(
            (const uint32_t*)w, wa);
        binlin_gemm_bf16<<<grid, 256, 0, stream>>>(
            (const __bf16*)xa, (const __bf16*)wa, bias, out);
    } else {
        binlin_wmma_fallback<<<grid, 256, 0, stream>>>(x, w, bias, out);
    }
}
